// UP2ME_forecaster_36721970381345
// MI455X (gfx1250) — compile-verified, matrix operand order checked by reading, not run on hardware
//
#include <hip/hip_runtime.h>
#include <hip/hip_bf16.h>
#include <stdint.h>

// ---------------------------------------------------------------------------
// UP2ME forecaster forward on gfx1250 (MI455X).
//  * All GEMMs -> bf16 WMMA (v_wmma_f32_16x16x32_bf16), f32 accumulate.
//  * Weight (B) tiles -> Tensor Data Mover (tensor_load_to_lds + TENSORcnt):
//    weights are pre-converted once per call to zero-padded transposed bf16
//    panels [Npad, K] so a 64x32 tile is a clean 2D strided TDM descriptor.
//  * Activation (A) tiles -> vectorized float4 loads + packed bf16 -> b64 LDS
//    stores on the full-K fast path.
//  * LN / softmax / top-k / embeds are wave32 VALU kernels (trivial next to
//    the ~1.3 TFLOP of matrix work; weights are L2-resident in 192MB L2).
// ---------------------------------------------------------------------------

typedef __attribute__((ext_vector_type(16))) __bf16 v16bf;
typedef __attribute__((ext_vector_type(8)))  float  v8f;
typedef __attribute__((ext_vector_type(4)))  unsigned u32x4_t;
typedef __attribute__((ext_vector_type(4)))  int      i32x4_t;
typedef __attribute__((ext_vector_type(8)))  int      i32x8_t;

#define ACT_NONE 0
#define ACT_GELU 1

// B-operand kinds
#define BK_F32_KN  0   // f32 [K,N] (needs transpose gather into LDS)
#define BK_F32_NK  1   // f32 [N,K] row-major ("transB": K^T, graph feat)
#define BK_BF16_NK 2   // bf16 [Npad,K] pre-transposed weight panel -> TDM

#if __has_builtin(__builtin_amdgcn_tensor_load_to_lds)
#define UP2ME_HAVE_TDM 1
#else
#define UP2ME_HAVE_TDM 0
#endif

// fp32 -> bf16 round-to-nearest-even bits (pure integer path)
__device__ __forceinline__ unsigned bfbits(float x) {
  union { float f; unsigned u; } cv; cv.f = x;
  return ((cv.u + 0x7FFFu + ((cv.u >> 16) & 1u)) >> 16) & 0xFFFFu;
}
__device__ __forceinline__ __bf16 f2bf(float x) {
  unsigned short h = (unsigned short)bfbits(x);
  return __builtin_bit_cast(__bf16, h);
}

__device__ __forceinline__ float gelu_f(float x) {
  float x3 = x * x * x;
  return 0.5f * x * (1.0f + tanhf(0.79788456080286535588f * (x + 0.044715f * x3)));
}

__device__ __forceinline__ float wave_sum(float v) {
  for (int m = 16; m > 0; m >>= 1) v += __shfl_xor(v, m, 32);
  return v;
}
__device__ __forceinline__ float wave_max(float v) {
  for (int m = 16; m > 0; m >>= 1) v = fmaxf(v, __shfl_xor(v, m, 32));
  return v;
}

// Gather one 16x32 bf16 WMMA fragment from an LDS row (row-major, 32 halves).
// Lane L(0-15): row M=L, K groups {0..7,16..23}; lanes 16-31: K groups
// {8..15,24..31}. Two contiguous 8-half runs -> 2x ds_load_b128.
__device__ __forceinline__ v16bf load_frag(const __bf16* row, int kbase) {
  v16bf f;
#pragma unroll
  for (int v = 0; v < 8; ++v) {
    int k0 = ((v & 4) ? 16 : 0) + kbase + 2 * (v & 3);
    f[2 * v]     = row[k0];
    f[2 * v + 1] = row[k0 + 1];
  }
  return f;
}

#if UP2ME_HAVE_TDM
// Issue one TDM 2D tile load: 64 rows x 32 bf16 (64B/row), row stride =
// ldbHalves elements, into LDS at ldsOff (contiguous 64x32 halves).
// D# packing per CDNA5 ISA ch.8: group0 = {count|flags, lds_addr, gaddr_lo,
// gaddr_hi|type=2}; group1 = {data_size=2B, tensor_dim0/1, tile_dim0/1/2,
// tensor_dim0_stride}. Groups 2/3 zero (2D tensor).
__device__ __forceinline__ void tdm_load_tile_b(const __bf16* gsrc,
                                                unsigned ldsOff, int ldbHalves) {
  unsigned long long ga = (unsigned long long)(uintptr_t)gsrc;
  u32x4_t g0;
  g0[0] = 1u;                                                  // count=1
  g0[1] = ldsOff;                                              // lds_addr
  g0[2] = (unsigned)(ga & 0xFFFFFFFFull);                      // gaddr[31:0]
  g0[3] = (unsigned)((ga >> 32) & 0x1FFFFFFull) | 0x80000000u; // [56:32]|type=2
  i32x8_t g1;
  g1[0] = 0x00010000;                   // workgroup_mask=0, data_size=1 (2B)
  g1[1] = (int)(32u << 16);             // tensor_dim0[15:0]=32  (bits 63:48)
  g1[2] = (int)(64u << 16);             // tensor_dim1[15:0]=64  (bits 95:80)
  g1[3] = (int)(32u << 16);             // tile_dim0=32          (bits 127:112)
  g1[4] = (int)(64u);                   // tile_dim1=64, tile_dim2=0
  g1[5] = ldbHalves;                    // tensor_dim0_stride[31:0]
  g1[6] = 0;                            // stride0[47:32]=0, dim1_stride lo=0
  g1[7] = 0;
  i32x4_t z4 = {0, 0, 0, 0};
#if defined(__clang_major__) && __clang_major__ >= 23
  i32x8_t z8 = {0, 0, 0, 0, 0, 0, 0, 0};
  __builtin_amdgcn_tensor_load_to_lds(g0, g1, z4, z4, z8, 0);
#else
  __builtin_amdgcn_tensor_load_to_lds(g0, g1, z4, z4, 0);
#endif
}
#endif  // UP2ME_HAVE_TDM

// ---------------------------------------------------------------------------
// Batched GEMM: C = act(alpha * A @ op(B) + bias)
//   A: f32 [M,K] rows at lda, batched: base += outer*sA0 + inner*sA1
//   B: per bkind (see BK_*); bf16 panels are zero-padded to Npad>=64 rows
//      and K multiple of 32 -> TDM tile loads need no guards.
//   C: f32 [M,N] rows at ldc, batched via sC0/sC1
// 128 threads (4 wave32), 64x64 macro-tile, each wave 32x32 via 2x2 WMMA.
// ---------------------------------------------------------------------------
__global__ void __launch_bounds__(128)
wmma_gemm_k(const float* __restrict__ A, const void* __restrict__ Bv,
            const float* __restrict__ bias, float* __restrict__ Cm,
            int M, int N, int K, int lda, int ldb, int ldc,
            long long sA0, long long sA1, long long sB0, long long sB1,
            long long sC0, long long sC1,
            int innerB, int bkind, float alpha, int act) {
  __shared__ __bf16 lA[64][32];
  __shared__ __bf16 lB[64][32];  // [n][k]

  int bz = blockIdx.y;
  int outer = bz / innerB, inner = bz % innerB;
  A += outer * sA0 + inner * sA1;
  Cm += outer * sC0 + inner * sC1;
  const float*  Bf = (const float*)Bv  + outer * sB0 + inner * sB1;
  const __bf16* Bt = (const __bf16*)Bv + outer * sB0 + inner * sB1;

  int tilesN = (N + 63) >> 6;
  int tileM = (int)blockIdx.x / tilesN;
  int tileN = (int)blockIdx.x % tilesN;

  int tid = threadIdx.x;
  int lane = tid & 31, wid = tid >> 5;
  int wM = (wid >> 1) * 32, wN = (wid & 1) * 32;
  int laneLo = lane & 15;
  int kbase = (lane >> 4) * 8;

  v8f acc[2][2] = {};

  for (int kt = 0; kt < K; kt += 32) {
    // ---- stage A tile (f32 -> bf16) ----
    if (kt + 32 <= K) {
      // fast path: vectorized float4 loads, packed bf16, b64 LDS stores
      for (int i = tid; i < 512; i += 128) {
        int r = i >> 3, kc = (i & 7) << 2;
        int gr = tileM * 64 + r;
        unsigned p0 = 0, p1 = 0;
        if (gr < M) {
          float4 v = *reinterpret_cast<const float4*>(
              A + (long long)gr * lda + kt + kc);
          p0 = bfbits(v.x) | (bfbits(v.y) << 16);
          p1 = bfbits(v.z) | (bfbits(v.w) << 16);
        }
        *reinterpret_cast<uint2*>(&lA[r][kc]) = make_uint2(p0, p1);
      }
    } else {
      for (int i = tid; i < 64 * 32; i += 128) {
        int r = i >> 5, k = i & 31;
        int gr = tileM * 64 + r, gk = kt + k;
        float v = (gr < M && gk < K) ? A[(long long)gr * lda + gk] : 0.f;
        lA[r][k] = f2bf(v);
      }
    }

    // ---- stage B tile ----
    if (bkind == BK_BF16_NK) {
#if UP2ME_HAVE_TDM
      if (tid < 32) {  // wave 0 only (TDM ignores EXEC; branch around it)
        tdm_load_tile_b(Bt + (size_t)(tileN * 64) * ldb + kt,
                        (unsigned)(uintptr_t)&lB[0][0], ldb);
        __builtin_amdgcn_s_wait_tensorcnt(0);
      }
#else
      for (int i = tid; i < 64 * 32; i += 128) {
        int n = i >> 5, k = i & 31;
        lB[n][k] = Bt[(size_t)(tileN * 64 + n) * ldb + kt + k];
      }
#endif
    } else {
      for (int i = tid; i < 64 * 32; i += 128) {
        int n = i >> 5, k = i & 31;
        int gn = tileN * 64 + n, gk = kt + k;
        float v = 0.f;
        if (gn < N && gk < K)
          v = (bkind == BK_F32_NK) ? Bf[(long long)gn * ldb + gk]
                                   : Bf[(long long)gk * ldb + gn];
        lB[n][k] = f2bf(v);
      }
    }
    __syncthreads();

    v16bf afr[2], bfr[2];
#pragma unroll
    for (int r = 0; r < 2; ++r)
      afr[r] = load_frag(&lA[wM + r * 16 + laneLo][0], kbase);
#pragma unroll
    for (int c = 0; c < 2; ++c)
      bfr[c] = load_frag(&lB[wN + c * 16 + laneLo][0], kbase);

#pragma unroll
    for (int r = 0; r < 2; ++r)
#pragma unroll
      for (int c = 0; c < 2; ++c)
        acc[r][c] = __builtin_amdgcn_wmma_f32_16x16x32_bf16(
            false, afr[r], false, bfr[c], (short)0, acc[r][c], false, false);
    __syncthreads();
  }

  // epilogue: C/D f32 layout — VGPR i: lanes 0-15 -> M=i, lanes 16-31 -> M=i+8
  int rowHi = (lane >= 16) ? 8 : 0;
#pragma unroll
  for (int r = 0; r < 2; ++r) {
#pragma unroll
    for (int c = 0; c < 2; ++c) {
#pragma unroll
      for (int i = 0; i < 8; ++i) {
        int row = tileM * 64 + wM + r * 16 + i + rowHi;
        int col = tileN * 64 + wN + c * 16 + laneLo;
        if (row < M && col < N) {
          float v = acc[r][c][i] * alpha + (bias ? bias[col] : 0.f);
          if (act == ACT_GELU) v = gelu_f(v);
          Cm[(long long)row * ldc + col] = v;
        }
      }
    }
  }
}

// W [K,N] f32 -> Wt [Npad,K] bf16, zero-padded rows (so TDM needs no guards)
__global__ void conv_wt_k(const float* __restrict__ W, __bf16* __restrict__ Wt,
                          int K, int N, long long total) {
  long long idx = (long long)blockIdx.x * blockDim.x + threadIdx.x;
  if (idx >= total) return;
  int k = (int)(idx % K);
  int n = (int)(idx / K);
  unsigned short h = 0;
  if (n < N) h = (unsigned short)bfbits(W[(long long)k * N + n]);
  Wt[idx] = __builtin_bit_cast(__bf16, h);
}

// out = LayerNorm(x + h) * g + beta.  One wave32 per row of D=256.
// If hTrans: x is [B,C,T,D] but h is [B,T,C,D] (channel-attn output layout).
__global__ void add_ln_k(const float* __restrict__ x, const float* __restrict__ h,
                         float* __restrict__ out,
                         const float* __restrict__ g, const float* __restrict__ be,
                         int Cc, int Tt, int Dd, int hTrans) {
  long long row = blockIdx.x;  // over B*C*T
  int t = (int)(row % Tt);
  long long bc = row / Tt;
  int c = (int)(bc % Cc);
  long long b = bc / Cc;
  const float* xp = x + row * Dd;
  const float* hp = hTrans ? h + (((b * Tt + t) * Cc) + c) * (long long)Dd
                           : h + row * Dd;
  int lane = threadIdx.x;
  float vals[8];
  int per = Dd >> 5;
  float s = 0.f;
#pragma unroll
  for (int i = 0; i < 8; ++i) {
    if (i < per) {
      float v = xp[lane + i * 32] + hp[lane + i * 32];
      vals[i] = v;
      s += v;
    }
  }
  s = wave_sum(s);
  float mu = s / (float)Dd;
  float var = 0.f;
#pragma unroll
  for (int i = 0; i < 8; ++i)
    if (i < per) { float d = vals[i] - mu; var += d * d; }
  var = wave_sum(var) / (float)Dd;
  float inv = rsqrtf(var + 1e-5f);
#pragma unroll
  for (int i = 0; i < 8; ++i) {
    if (i < per) {
      int d = lane + i * 32;
      out[row * Dd + d] = (vals[i] - mu) * inv * g[d] + be[d];
    }
  }
}

// Row softmax (one wave32 per row, rowLen <= 128). Optional adjacency mask:
// rows laid out as ((b*T + t)*H + h)*M + i ; mask row = adj[b][i][*] over Cc.
__global__ void softmax_rows_k(float* __restrict__ S, int rowLen,
                               const float* __restrict__ adj,
                               int perB, int Mm, int Cc) {
  long long row = blockIdx.x;
  float* p = S + row * rowLen;
  const float* mrow = nullptr;
  if (adj) {
    int b = (int)(row / perB);
    int i = (int)(row % Mm);
    mrow = adj + ((long long)b * Cc + i) * Cc;
  }
  int lane = threadIdx.x;
  float vals[4];
  float mx = -3.0e38f;
  int cnt = 0;
  for (int j = lane; j < rowLen; j += 32) {
    float v = p[j];
    if (mrow && mrow[j] < 0.5f) v = -1.0e9f;
    vals[cnt++] = v;
    mx = fmaxf(mx, v);
  }
  mx = wave_max(mx);
  float sum = 0.f;
  cnt = 0;
  for (int j = lane; j < rowLen; j += 32) {
    float e = __expf(vals[cnt] - mx);
    vals[cnt++] = e;
    sum += e;
  }
  sum = wave_sum(sum);
  float inv = 1.f / sum;
  cnt = 0;
  for (int j = lane; j < rowLen; j += 32) p[j] = vals[cnt++] * inv;
}

// x = patch_embed(ts patches) + pos + chan
__global__ void patch_embed_k(const float* __restrict__ ts,
                              const float* __restrict__ W, const float* __restrict__ bias,
                              const float* __restrict__ pos, const float* __restrict__ chan,
                              float* __restrict__ out,
                              int Cc, int Tt, int P, int Dd, long long total) {
  long long idx = (long long)blockIdx.x * blockDim.x + threadIdx.x;
  if (idx >= total) return;
  int d = (int)(idx % Dd);
  long long r = idx / Dd;
  int t = (int)(r % Tt);
  long long bc = r / Tt;
  int c = (int)(bc % Cc);
  const float* pt = ts + bc * (long long)(Tt * P) + (long long)t * P;
  float acc = bias[d];
#pragma unroll
  for (int p = 0; p < 12; ++p) acc += pt[p] * W[p * Dd + d];
  out[idx] = acc + pos[t * Dd + d] + chan[c * Dd + d];
}

// normalize rows to unit L2 (+1e-8)
__global__ void row_normalize_k(const float* __restrict__ in, float* __restrict__ out,
                                int len) {
  long long row = blockIdx.x;
  const float* p = in + row * (long long)len;
  float s = 0.f;
  for (int i = threadIdx.x; i < len; i += blockDim.x) { float v = p[i]; s += v * v; }
  __shared__ float red[256];
  red[threadIdx.x] = s;
  __syncthreads();
  for (int st = 128; st > 0; st >>= 1) {
    if ((int)threadIdx.x < st) red[threadIdx.x] += red[threadIdx.x + st];
    __syncthreads();
  }
  float inv = 1.f / (sqrtf(red[0]) + 1e-8f);
  float* q = out + row * (long long)len;
  for (int i = threadIdx.x; i < len; i += blockDim.x) q[i] = p[i] * inv;
}

// top-k neighbor selection per (b,i) row of sim, writes 0/1 row
__global__ void topk_adj_k(const float* __restrict__ sim, float* __restrict__ adjraw,
                           int Bb, int Cc, int k) {
  int id = blockIdx.x * blockDim.x + threadIdx.x;
  if (id >= Bb * Cc) return;
  const float* row = sim + (long long)id * Cc;
  float* arow = adjraw + (long long)id * Cc;
  for (int j = 0; j < Cc; ++j) arow[j] = 0.f;
  unsigned long long chosen = 0ull;
  for (int t = 0; t < k; ++t) {
    float best = -3.0e38f;
    int bj = 0;
    for (int j = 0; j < Cc; ++j) {
      if (chosen & (1ull << j)) continue;
      float v = row[j];
      if (v > best) { best = v; bj = j; }
    }
    chosen |= (1ull << bj);
    arow[bj] = 1.f;
  }
}

__global__ void sym_adj_k(const float* __restrict__ raw, float* __restrict__ adj,
                          int Cc, long long total) {
  long long id = (long long)blockIdx.x * blockDim.x + threadIdx.x;
  if (id >= total) return;
  long long b = id / ((long long)Cc * Cc);
  int r = (int)(id % ((long long)Cc * Cc));
  int i = r / Cc, j = r % Cc;
  long long base = b * Cc * Cc;
  adj[id] = fmaxf(raw[base + (long long)i * Cc + j], raw[base + (long long)j * Cc + i]);
}

// X[seq, Tpast+t, :] = learnable + pos[Tpast+t] + chan[seq%C]
__global__ void fut_fill_k(float* __restrict__ X, const float* __restrict__ lp,
                           const float* __restrict__ pos, const float* __restrict__ chan,
                           int Cc, int Tpast, int Tpred, int Dd, long long total) {
  long long idx = (long long)blockIdx.x * blockDim.x + threadIdx.x;
  if (idx >= total) return;
  int d = (int)(idx % Dd);
  long long r = idx / Dd;
  int t = (int)(r % Tpred);
  long long seq = r / Tpred;
  int c = (int)(seq % Cc);
  X[(seq * (Tpast + Tpred) + Tpast + t) * Dd + d] =
      lp[d] + pos[(Tpast + t) * Dd + d] + chan[c * Dd + d];
}

__global__ void chan_add_k(float* __restrict__ X, const float* __restrict__ chan,
                           int Cc, int Tt, int Dd, long long total) {
  long long idx = (long long)blockIdx.x * blockDim.x + threadIdx.x;
  if (idx >= total) return;
  int d = (int)(idx % Dd);
  long long seq = idx / ((long long)Tt * Dd);
  int c = (int)(seq % Cc);
  X[idx] += chan[c * Dd + d];
}

// d_out[seq, k] = OutB[seq, (full - tail) + k]
__global__ void slice_out_k(const float* __restrict__ OutB, float* __restrict__ out,
                            int tail, int full, long long total) {
  long long idx = (long long)blockIdx.x * blockDim.x + threadIdx.x;
  if (idx >= total) return;
  int k = (int)(idx % tail);
  long long seq = idx / tail;
  out[idx] = OutB[seq * full + (full - tail) + k];
}

// ---------------------------------------------------------------------------
// Host-side orchestration
// ---------------------------------------------------------------------------
struct Lin { const float* W; const float* b; __bf16* Wt; int K, N; };
struct LnP { const float *g, *b; };
struct AttnP { Lin q, k, v, o; };
struct EncL { AttnP attn; Lin ff1, ff2; LnP ln1, ln2; };
struct TcL  { AttnP t, c; Lin ff1, ff2; LnP ln1, ln2, ln3; };

static inline void launch_gemm(hipStream_t st, const float* A, const void* B,
                               const float* bias, float* C,
                               int M, int N, int K, int lda, int ldb, int ldc,
                               long long sA0, long long sA1,
                               long long sB0, long long sB1,
                               long long sC0, long long sC1,
                               int innerB, int nBatch, int bkind,
                               float alpha, int act) {
  int tm = (M + 63) / 64, tn = (N + 63) / 64;
  dim3 grid((unsigned)(tm * tn), (unsigned)nBatch);
  wmma_gemm_k<<<grid, dim3(128), 0, st>>>(A, B, bias, C, M, N, K, lda, ldb, ldc,
                                          sA0, sA1, sB0, sB1, sC0, sC1,
                                          innerB, bkind, alpha, act);
}

// GEMM against a pre-transposed bf16 weight panel (TDM path)
static inline void gemm_w(hipStream_t st, const float* A, const Lin& w, float* C,
                          int M, int ldaA, long long sA0, long long sA1,
                          long long sC0, long long sC1, int innerB, int nBatch,
                          int ldc, int act) {
  launch_gemm(st, A, w.Wt, w.b, C, M, w.N, w.K, ldaA, w.K, ldc,
              sA0, sA1, 0, 0, sC0, sC1, innerB, nBatch, BK_BF16_NK, 1.f, act);
}

static void run_temporal_mha(hipStream_t st, const float* Xb, const AttnP& w,
                             int nSeq, int L,
                             float* Q, float* Kb, float* V, float* S,
                             float* O, float* Ht) {
  const int D = 256, H = 4;
  long long sx = (long long)L * D;
  gemm_w(st, Xb, w.q, Q,  L, D, sx, 0, sx, 0, 1, nSeq, D, ACT_NONE);
  gemm_w(st, Xb, w.k, Kb, L, D, sx, 0, sx, 0, 1, nSeq, D, ACT_NONE);
  gemm_w(st, Xb, w.v, V,  L, D, sx, 0, sx, 0, 1, nSeq, D, ACT_NONE);
  long long sS = (long long)L * L;
  // scores = (Q_h @ K_h^T) / 8 ; batch = (seq, head)
  launch_gemm(st, Q, Kb, nullptr, S, L, L, 64, D, D, L,
              sx, 64, sx, 64, H * sS, sS, H, nSeq * H, BK_F32_NK, 0.125f, ACT_NONE);
  softmax_rows_k<<<dim3((unsigned)(nSeq * H * L)), dim3(32), 0, st>>>(S, L, nullptr, 1, 1, 1);
  // O_h = P @ V_h
  launch_gemm(st, S, V, nullptr, O, L, 64, L, L, D, D,
              H * sS, sS, sx, 64, sx, 64, H, nSeq * H, BK_F32_KN, 1.f, ACT_NONE);
  gemm_w(st, O, w.o, Ht, L, D, sx, 0, sx, 0, 1, nSeq, D, ACT_NONE);
}

static void run_ffn_ln(hipStream_t st, float* Xb, const Lin& f1, const Lin& f2,
                       const LnP& l, int nSeq, int L, int Cc,
                       float* Fb, float* Ht) {
  const int D = 256, DFF = 512;
  long long sx = (long long)L * D, sf = (long long)L * DFF;
  gemm_w(st, Xb, f1, Fb, L, D,   sx, 0, sf, 0, 1, nSeq, DFF, ACT_GELU);
  gemm_w(st, Fb, f2, Ht, L, DFF, sf, 0, sx, 0, 1, nSeq, D,   ACT_NONE);
  add_ln_k<<<dim3((unsigned)(nSeq * L)), dim3(32), 0, st>>>(Xb, Ht, Xb, l.g, l.b, Cc, L, D, 0);
}

static void run_transformer_layer(hipStream_t st, float* Xb, const EncL& e,
                                  int nSeq, int L, int Cc,
                                  float* Q, float* Kb, float* V, float* S,
                                  float* O, float* Ht, float* Fb) {
  run_temporal_mha(st, Xb, e.attn, nSeq, L, Q, Kb, V, S, O, Ht);
  add_ln_k<<<dim3((unsigned)(nSeq * L)), dim3(32), 0, st>>>(Xb, Ht, Xb, e.ln1.g, e.ln1.b, Cc, L, 256, 0);
  run_ffn_ln(st, Xb, e.ff1, e.ff2, e.ln2, nSeq, L, Cc, Fb, Ht);
}

extern "C" void kernel_launch(void* const* d_in, const int* in_sizes, int n_in,
                              void* d_out, int out_size, void* d_ws, size_t ws_size,
                              hipStream_t stream) {
  (void)in_sizes; (void)n_in; (void)out_size; (void)ws_size;
  constexpr int Bn = 16, Cn = 64, Dn = 256, Hn = 4;
  constexpr int PATCH = 12, Tpast = 96, Tpred = 24, Tfull = 120, NEIGH = 10;
  constexpr int nSeq = Bn * Cn;  // 1024

  // ---- unpack params (setup_inputs dict order, depth-first) ----
  Lin* lins[72]; int nLins = 0;
  int ix = 0;
  auto Fp  = [&]() { return (const float*)d_in[ix++]; };
  auto lnp = [&]() { LnP l; l.g = Fp(); l.b = Fp(); return l; };
  auto lin = [&](int K, int N) {
    Lin l; l.W = Fp(); l.b = Fp(); l.Wt = nullptr; l.K = K; l.N = N;
    return l;
  };
  auto reg = [&](Lin& l) { lins[nLins++] = &l; };

  const float* past_ts = Fp();
  Lin pe = lin(PATCH, Dn);               // used by dedicated kernel, no panel
  const float* pos  = Fp();
  const float* chan = Fp();
  auto attn = [&]() {
    AttnP a; a.q = lin(Dn, Dn); a.k = lin(Dn, Dn); a.v = lin(Dn, Dn); a.o = lin(Dn, Dn);
    return a;
  };
  EncL enc[4];
  for (auto& e : enc) {
    e.attn = attn(); e.ff1 = lin(Dn, 512); e.ff2 = lin(512, Dn);
    e.ln1 = lnp(); e.ln2 = lnp();
    reg(e.attn.q); reg(e.attn.k); reg(e.attn.v); reg(e.attn.o); reg(e.ff1); reg(e.ff2);
  }
  Lin e2d = lin(Dn, Dn); reg(e2d);
  const float* lpatch = Fp();
  TcL tcl[3];
  for (auto& e : tcl) {
    e.t = attn(); e.c = attn(); e.ff1 = lin(Dn, 512); e.ff2 = lin(512, Dn);
    e.ln1 = lnp(); e.ln2 = lnp(); e.ln3 = lnp();
    reg(e.t.q); reg(e.t.k); reg(e.t.v); reg(e.t.o);
    reg(e.c.q); reg(e.c.k); reg(e.c.v); reg(e.c.o);
    reg(e.ff1); reg(e.ff2);
  }
  EncL dec;
  dec.attn = attn(); dec.ff1 = lin(Dn, 512); dec.ff2 = lin(512, Dn);
  dec.ln1 = lnp(); dec.ln2 = lnp();
  reg(dec.attn.q); reg(dec.attn.k); reg(dec.attn.v); reg(dec.attn.o);
  reg(dec.ff1); reg(dec.ff2);
  Lin oproj = lin(Dn, PATCH); reg(oproj);
  // pred_patch_num / neighbor_num device scalars: compile-time constants here.

  // ---- workspace carve ----
  char* wsp = (char*)d_ws;
  auto carveB = [&](size_t bytes) {
    char* p = wsp;
    wsp += (bytes + 255) & ~(size_t)255;
    return p;
  };
  auto carve = [&](long long elems) { return (float*)carveB((size_t)elems * 4); };

  // bf16 transposed weight panels (zero-padded rows -> guard-free TDM tiles)
  for (int i = 0; i < nLins; ++i) {
    Lin& l = *lins[i];
    int Npad = (l.N + 63) & ~63;
    l.Wt = (__bf16*)carveB((size_t)Npad * l.K * sizeof(__bf16));
    long long tot = (long long)Npad * l.K;
    conv_wt_k<<<dim3((unsigned)((tot + 255) / 256)), dim3(256), 0, stream>>>(
        l.W, l.Wt, l.K, l.N, tot);
  }

  float* XE   = carve((long long)nSeq * Tpast * Dn);
  float* X    = carve((long long)nSeq * Tfull * Dn);
  float* Q    = carve((long long)nSeq * Tfull * Dn);
  float* Kb   = carve((long long)nSeq * Tfull * Dn);
  float* V    = carve((long long)nSeq * Tfull * Dn);
  float* O    = carve((long long)nSeq * Tfull * Dn);
  float* Ht   = carve((long long)nSeq * Tfull * Dn);
  float* Fb   = carve((long long)nSeq * Tfull * 512);
  float* S    = carve((long long)nSeq * Hn * Tfull * Tfull);
  float* sim  = carve((long long)Bn * Cn * Cn);
  float* araw = carve((long long)Bn * Cn * Cn);
  float* adj  = carve((long long)Bn * Cn * Cn);
  float* OutB = carve((long long)nSeq * Tfull * PATCH);

  // ---- 1. patch embed + pos + channel ----
  {
    long long tot = (long long)nSeq * Tpast * Dn;
    patch_embed_k<<<dim3((unsigned)((tot + 255) / 256)), dim3(256), 0, stream>>>(
        past_ts, pe.W, pe.b, pos, chan, XE, Cn, Tpast, PATCH, Dn, tot);
  }

  // ---- 2. encoder x4 ----
  for (int l = 0; l < 4; ++l)
    run_transformer_layer(stream, XE, enc[l], nSeq, Tpast, Cn, Q, Kb, V, S, O, Ht, Fb);

  // ---- 3. graph construction ----
  {
    const int FL = Tpast * Dn;  // 24576
    float* FN = Q;              // Q free here
    row_normalize_k<<<dim3((unsigned)nSeq), dim3(256), 0, stream>>>(XE, FN, FL);
    launch_gemm(stream, FN, FN, nullptr, sim, Cn, Cn, FL, FL, FL, Cn,
                (long long)Cn * FL, 0, (long long)Cn * FL, 0,
                (long long)Cn * Cn, 0, 1, Bn, BK_F32_NK, 1.f, ACT_NONE);
    topk_adj_k<<<dim3((unsigned)((Bn * Cn + 255) / 256)), dim3(256), 0, stream>>>(
        sim, araw, Bn, Cn, NEIGH);
    long long tot = (long long)Bn * Cn * Cn;
    sym_adj_k<<<dim3((unsigned)((tot + 255) / 256)), dim3(256), 0, stream>>>(araw, adj, Cn, tot);
  }

  // ---- 4. enc_2_dec into X rows [0,96), future fill rows [96,120) ----
  gemm_w(stream, XE, e2d, X, Tpast, Dn, (long long)Tpast * Dn, 0,
         (long long)Tfull * Dn, 0, 1, nSeq, Dn, ACT_NONE);
  {
    long long tot = (long long)nSeq * Tpred * Dn;
    fut_fill_k<<<dim3((unsigned)((tot + 255) / 256)), dim3(256), 0, stream>>>(
        X, lpatch, pos, chan, Cn, Tpast, Tpred, Dn, tot);
  }

  // ---- 5. temporal-channel layers x3 ----
  for (int l = 0; l < 3; ++l) {
    const TcL& p = tcl[l];
    run_temporal_mha(stream, X, p.t, nSeq, Tfull, Q, Kb, V, S, O, Ht);
    add_ln_k<<<dim3((unsigned)(nSeq * Tfull)), dim3(32), 0, stream>>>(
        X, Ht, X, p.ln1.g, p.ln1.b, Cn, Tfull, Dn, 0);

    // channel attention (rows = channels; strided A avoids transposes)
    {
      const AttnP& w = p.c;
      int ldaC = Tfull * Dn;                       // row stride over c in X
      long long sAo = (long long)Cn * Tfull * Dn;  // per b
      long long sAi = Dn;                          // per t
      long long sCo = (long long)Tfull * Cn * Dn;  // per b (channel-major out)
      long long sCi = (long long)Cn * Dn;          // per t
      gemm_w(stream, X, w.q, Q,  Cn, ldaC, sAo, sAi, sCo, sCi, Tfull, Bn * Tfull, Dn, ACT_NONE);
      gemm_w(stream, X, w.k, Kb, Cn, ldaC, sAo, sAi, sCo, sCi, Tfull, Bn * Tfull, Dn, ACT_NONE);
      gemm_w(stream, X, w.v, V,  Cn, ldaC, sAo, sAi, sCo, sCi, Tfull, Bn * Tfull, Dn, ACT_NONE);
      long long sS = (long long)Cn * Cn;
      long long so = (long long)Cn * Dn;  // per (b,t) in channel-major Q/K/V
      launch_gemm(stream, Q, Kb, nullptr, S, Cn, Cn, 64, Dn, Dn, Cn,
                  so, 64, so, 64, Hn * sS, sS, Hn, Bn * Tfull * Hn, BK_F32_NK, 0.125f, ACT_NONE);
      softmax_rows_k<<<dim3((unsigned)(Bn * Tfull * Hn * Cn)), dim3(32), 0, stream>>>(
          S, Cn, adj, Tfull * Hn * Cn, Cn, Cn);
      launch_gemm(stream, S, V, nullptr, O, Cn, 64, Cn, Cn, Dn, Dn,
                  Hn * sS, sS, so, 64, so, 64, Hn, Bn * Tfull * Hn, BK_F32_KN, 1.f, ACT_NONE);
      gemm_w(stream, O, w.o, Ht, Cn, Dn, so, 0, so, 0, 1, Bn * Tfull, Dn, ACT_NONE);
      // Ht is [B,T,C,D]; add_ln remaps it back onto X's [B,C,T,D]
      add_ln_k<<<dim3((unsigned)(nSeq * Tfull)), dim3(32), 0, stream>>>(
          X, Ht, X, p.ln2.g, p.ln2.b, Cn, Tfull, Dn, 1);
    }

    run_ffn_ln(stream, X, p.ff1, p.ff2, p.ln3, nSeq, Tfull, Cn, Fb, Ht);
  }

  // ---- 6. decoder: + channel embed, 1 transformer layer ----
  {
    long long tot = (long long)nSeq * Tfull * Dn;
    chan_add_k<<<dim3((unsigned)((tot + 255) / 256)), dim3(256), 0, stream>>>(
        X, chan, Cn, Tfull, Dn, tot);
  }
  run_transformer_layer(stream, X, dec, nSeq, Tfull, Cn, Q, Kb, V, S, O, Ht, Fb);

  // ---- 7. output projection + final slice ----
  gemm_w(stream, X, oproj, OutB, Tfull, Dn, (long long)Tfull * Dn, 0,
         (long long)Tfull * PATCH, 0, 1, nSeq, PATCH, ACT_NONE);
  {
    int tail = Tpred * PATCH;   // 288
    int full = Tfull * PATCH;   // 1440
    long long tot = (long long)nSeq * tail;
    slice_out_k<<<dim3((unsigned)((tot + 255) / 256)), dim3(256), 0, stream>>>(
        OutB, (float*)d_out, tail, full, tot);
  }
}
